// AdversarialMorphingLoss_48833778156001
// MI455X (gfx1250) — compile-verified, matrix operand order checked by reading, not run on hardware
//
#include <hip/hip_runtime.h>

typedef __attribute__((ext_vector_type(8))) int v8i;

#define BATCH 4096
#define SEQ   2048
#define TILES (BATCH/16)   // 256 blocks, one 16-flow tile each
#define WAVES 8            // 8 wave32 per block
#define SEG   (SEQ/WAVES)  // 256 packets per wave per row
#define ITERS (SEG/64)     // 4 WMMA steps (K=64 each) per wave

__global__ __launch_bounds__(256)
void morph_loss_main(const int*   __restrict__ sizes,
                     const float* __restrict__ delays,
                     const int*   __restrict__ dirs,
                     const float* __restrict__ delay_ms,
                     const float* __restrict__ padding_norm,
                     const float* __restrict__ confidence,
                     const int*   __restrict__ profile_ids,
                     float*       __restrict__ partials)
{
    __shared__ int   lds_rows[WAVES][16];
    __shared__ float lds_terms[16];

    const int tile = blockIdx.x;
    const int tid  = threadIdx.x;
    const int wave = tid >> 5;
    const int lane = tid & 31;
    const int mrow = lane & 15;      // WMMA A-matrix row (lanes L and L+16 -> M=L)
    const int half = lane >> 4;      // which 32-packet half of the K=64 slice
    const int row  = tile * 16 + mrow;
    const size_t base_row = (size_t)row * SEQ;

    // per-row morphing parameters (only consumed for the global last packet)
    const float padv = padding_norm[row];
    const float dlyv = delay_ms[row];

    v8i acc;
    v8i bones;
#pragma unroll
    for (int q = 0; q < 8; ++q) { acc[q] = 0; bones[q] = 0x01010101; }

    for (int it = 0; it < ITERS; ++it) {
        const int p0 = wave * SEG + it * 64 + half * 32;   // 128B-aligned chunk start
        const int*   sp = sizes  + base_row + p0;
        const float* dp = delays + base_row + p0;
        const int*   rp = dirs   + base_row + p0;

        if (it + 1 < ITERS) {  // global_prefetch_b8 for the next chunk
            __builtin_prefetch((const void*)(sp + 64), 0, 1);
            __builtin_prefetch((const void*)(dp + 64), 0, 1);
            __builtin_prefetch((const void*)(rp + 64), 0, 1);
        }

        // previous-packet state at chunk entry (raw values; sentinel for j==0)
        float prevS, prevD;
        if (p0 == 0) { prevS = -1.0f; prevD = -1.0f; }
        else         { prevS = (float)sp[-1]; prevD = (float)rp[-1]; }

        v8i a;
#pragma unroll
        for (int q = 0; q < 8; ++q) {
            const int4   sv = ((const int4*)sp)[q];
            const float4 dv = ((const float4*)dp)[q];
            const int4   rv = ((const int4*)rp)[q];
            const int   svA[4] = {sv.x, sv.y, sv.z, sv.w};
            const float dvA[4] = {dv.x, dv.y, dv.z, dv.w};
            const int   rvA[4] = {rv.x, rv.y, rv.z, rv.w};
            unsigned packed = 0u;
#pragma unroll
            for (int j = 0; j < 4; ++j) {
                const int  gj   = p0 + q * 4 + j;
                const bool last = (gj == SEQ - 1);
                float sz = (float)svA[j];
                float dl = dvA[j];
                if (last) { sz = fminf(sz + padv * 1500.0f, 1500.0f); dl += dlyv; }
                const float dr = (float)rvA[j];
                unsigned sc = 0u;                       // 10x-integer suspicion score
                sc += (sz > 1400.0f)              ? 6u : 0u;
                sc += (dl < 0.05f)                ? 4u : 0u;
                sc += (fabsf(sz - prevS) < 0.5f)  ? 2u : 0u;
                sc += (dr != prevD)               ? 1u : 0u;
                packed |= sc << (8 * j);
                prevS = (float)svA[j];                  // prev is always the raw value
                prevD = dr;
            }
            a[q] = (int)packed;
        }
        // D[m,n] = sum_k A[m,k] : byte row-sum via IU8 WMMA with all-ones B
        acc = __builtin_amdgcn_wmma_i32_16x16x64_iu8(false, a, false, bones, acc,
                                                     false, false);
    }

    // C/D layout: VGPR r, lane 0 -> (M=r, N=0); lane 16 -> (M=8+r, N=0)
    if (lane == 0) {
#pragma unroll
        for (int r = 0; r < 8; ++r) lds_rows[wave][r] = acc[r];
    } else if (lane == 16) {
#pragma unroll
        for (int r = 0; r < 8; ++r) lds_rows[wave][8 + r] = acc[r];
    }
    __syncthreads();

    if (tid < 16) {
        int rs = 0;
#pragma unroll
        for (int w = 0; w < WAVES; ++w) rs += lds_rows[w][tid];

        const int   grow = tile * 16 + tid;
        const int   pid  = profile_ids[grow];
        const float CMULT[4] = {1.0f, 1.3f, 1.6f, 2.0f};
        const float TD[5]    = {2.0f, 1.0f, 0.5f, 5.0f, 3.0f};
        const float TP[5]    = {0.08f, 0.12f, 0.05f, 0.15f, 0.10f};

        const float score = (float)rs * (10.0f / (float)SEQ) * CMULT[pid & 3];
        const float dly   = delay_ms[grow];
        const float pad   = padding_norm[grow];
        const float conf  = confidence[grow];

        const float dpi   = fmaxf(score - 15.0f, 0.0f) * (1.0f / 30.0f);
        const float sim   = fabsf(dly - TD[pid]) + fabsf(pad - TP[pid]);
        const float eff   = fmaxf(dly - 20.0f, 0.0f) * (1.0f / 20.0f)
                          + fmaxf(pad - 0.3f, 0.0f);
        const float evade = (score < 30.0f) ? 1.0f : 0.0f;
        const float cl    = (conf - evade) * (conf - evade);

        lds_terms[tid] = 2.0f * dpi + 0.5f * sim + 0.3f * eff + 0.2f * cl;
    }
    __syncthreads();

    if (tid == 0) {
        float s = 0.0f;
#pragma unroll
        for (int t = 0; t < 16; ++t) s += lds_terms[t];
        partials[tile] = s * (1.0f / (float)BATCH);
    }
}

__global__ void morph_loss_finalize(const float* __restrict__ partials,
                                    float* __restrict__ out)
{
    if (threadIdx.x == 0 && blockIdx.x == 0) {
        float s = 0.0f;
        for (int i = 0; i < TILES; ++i) s += partials[i];   // fixed order: deterministic
        out[0] = s;
    }
}

extern "C" void kernel_launch(void* const* d_in, const int* in_sizes, int n_in,
                              void* d_out, int out_size, void* d_ws, size_t ws_size,
                              hipStream_t stream)
{
    const int*   sizes  = (const int*)  d_in[0];
    const float* delays = (const float*)d_in[1];
    const int*   dirs   = (const int*)  d_in[2];
    const float* dly    = (const float*)d_in[3];
    const float* pad    = (const float*)d_in[4];
    const float* conf   = (const float*)d_in[5];
    const int*   pid    = (const int*)  d_in[6];

    float* out      = (float*)d_out;
    float* partials = (float*)d_ws;    // 256 floats of scratch

    morph_loss_main<<<TILES, 256, 0, stream>>>(sizes, delays, dirs,
                                               dly, pad, conf, pid, partials);
    morph_loss_finalize<<<1, 32, 0, stream>>>(partials, out);
}
